// PKSpellHierarchical_89816356094113
// MI455X (gfx1250) — matrix operand order repaired and checked by reading, not compiled, not run on hardware
//
#include <hip/hip_runtime.h>

// ---------------------------------------------------------------------------
// PKSpellHierarchical loss on MI455X (gfx1250, wave32, WMMA bf16/f32-acc).
//
// Assumed d_in flattening (insertion order of setup_inputs() dict, params
// nested dicts recursed in insertion order):
//  0 sentences[T,B,17] f32
//  rnn  : 1 Wih_f 2 Whh_f 3 bih_f 4 bhh_f 5 Wih_b 6 Whh_b 7 bih_b 8 bhh_b
//  rnn2 : 9..16   hier1: 17..24   hier2: 25..32
//  att1 : 33 Wq 34 bq 35 Wv 36 bv   att2: 37..40
//  41 Wp 42 bp 43 Wk 44 bk
//  45 pitches[T,B] i32  46 keysignatures[T,B] i32  47 sentences_len  48 eoM
// ---------------------------------------------------------------------------

typedef __attribute__((ext_vector_type(16))) __bf16 v16bf;
typedef __attribute__((ext_vector_type(8)))  __bf16 v8bf;
typedef __attribute__((ext_vector_type(8)))  float  v8f;
typedef __attribute__((ext_vector_type(4)))  float  v4f;

#define T_LEN 1024
#define NB    96
#define DIN   17
#define H1    150
#define G1    450           // 3*H1
#define G1P   464           // 29*16 padded gate columns
#define NT1   29            // N tiles of 16 covering 450
#define H2    12
#define G2    36
#define G2P   48
#define HH    16
#define GH_   48
#define SEG   16
#define NSEG  64
#define BN    6144          // NB*NSEG
#define NP    35
#define NK    15
#define DPC   332           // 300 + 32
#define DKC   56            // 24 + 32

// ------------------------------ WMMA helpers -------------------------------

static __device__ inline v8f wmma_bf(v16bf a, v16bf b, v8f c) {
  return __builtin_amdgcn_wmma_f32_16x16x32_bf16(false, a, false, b, (short)0, c,
                                                 false, false);
}

// A-operand K index pattern (16-bit A 16x32): lanes 0-15 hold K{0..7,16..23},
// lanes 16-31 hold K{8..15,24..31}.
static __device__ inline int a_kidx(int i, int hi) {
  return ((i >> 3) << 4) + (i & 7) + (hi << 3);
}

// A tile from LDS bf16 row-major [rows][ld]  (lane-contiguous -> ds_load_b128)
static __device__ inline v16bf load_a_lds(const __bf16* base, int row0, int ld,
                                          int kbase, int lane) {
  int m = row0 + (lane & 15);
  int hi = (lane >> 4) & 1;
  v16bf a;
#pragma unroll
  for (int i = 0; i < 16; ++i) a[i] = base[m * ld + kbase + a_kidx(i, hi)];
  return a;
}

// A tile from LDS f32 row-major [rows][ld] with K clamp, converting to bf16
static __device__ inline v16bf load_a_ldsf32(const float* base, int row0, int ld,
                                             int ktot, int lane) {
  int m = row0 + (lane & 15);
  int hi = (lane >> 4) & 1;
  v16bf a;
#pragma unroll
  for (int i = 0; i < 16; ++i) {
    int k = a_kidx(i, hi);
    a[i] = (k < ktot) ? (__bf16)base[m * ld + k] : (__bf16)0.0f;
  }
  return a;
}

// A tile from global bf16 [.][ld] with K clamp (pad-zero)
static __device__ inline v16bf load_a_gbf(const __bf16* X, int row0, int ld,
                                          int kbase, int ktot, int lane) {
  int m = row0 + (lane & 15);
  int hi = (lane >> 4) & 1;
  v16bf a;
#pragma unroll
  for (int i = 0; i < 16; ++i) {
    int k = kbase + a_kidx(i, hi);
    a[i] = (k < ktot) ? X[m * ld + k] : (__bf16)0.0f;
  }
  return a;
}

// B tile (Kx16) built from row-major weight W[N][ldw]:  B[k][n] = W[n][k0+k].
// Lane layout: n = lane&15, lanes 0-15 hold K{0..7,16..23}, lanes 16-31 hold K{8..15,24..31}.
static __device__ inline v16bf load_b_w(const float* W, int n0, int ntot,
                                        int k0, int ktot, int ldw, int lane) {
  int n = n0 + (lane & 15);
  int hi = (lane >> 4) & 1;
  int kb = k0;
  v16bf b;
#pragma unroll
  for (int i = 0; i < 16; ++i) {
    int k = kb + a_kidx(i, hi);
    float v = (n < ntot && k < ktot) ? W[n * ldw + k] : 0.0f;
    b[i] = (__bf16)v;
  }
  return b;
}

// C/D tile stores into COLUMN-MAJOR gate scratch [cols][ldr rows]: a lane's 8
// accumulator values are consecutive rows of one column -> single ds_store_b128.
static __device__ inline void store_c_col_bf(__bf16* dst, int ldr, int col0,
                                             int row0, v8f acc, int lane) {
  int n = col0 + (lane & 15);
  int mb = row0 + ((lane >> 4) & 1) * 8;
  v8bf p;
#pragma unroll
  for (int v = 0; v < 8; ++v) p[v] = (__bf16)acc[v];
  *(v8bf*)(dst + n * ldr + mb) = p;
}

static __device__ inline void store_c_col_f32(float* dst, int ldr, int col0,
                                              int row0, v8f acc, int lane) {
  int n = col0 + (lane & 15);
  int mb = row0 + ((lane >> 4) & 1) * 8;
  v4f lo, hi2;
#pragma unroll
  for (int v = 0; v < 4; ++v) { lo[v] = acc[v]; hi2[v] = acc[4 + v]; }
  *(v4f*)(dst + n * ldr + mb) = lo;
  *(v4f*)(dst + n * ldr + mb + 4) = hi2;
}

static __device__ inline float sigm(float x) { return 1.0f / (1.0f + __expf(-x)); }

// ----------------------- CDNA5 async global->LDS copy ----------------------
// GVS form: per-lane LDS byte address (VDST), uniform SGPR base, per-lane
// 32-bit byte offset (VADDR).  Tracked by ASYNCcnt.
static __device__ inline void async_g2l_b32(unsigned lds_byte, unsigned long long gbase,
                                            int voff_bytes) {
  asm volatile("global_load_async_to_lds_b32 %0, %1, %2"
               :
               : "v"(lds_byte), "v"(voff_bytes), "s"(gbase)
               : "memory");
}
static __device__ inline void wait_asynccnt0() {
  asm volatile("s_wait_asynccnt 0x0" ::: "memory");
}
static __device__ inline unsigned lds_addr32(const void* p) {
  return (unsigned)(unsigned long long)(size_t)p;   // low 32 bits = LDS offset
}

// --------------------------- main biGRU (H1=150) ---------------------------
// blockIdx.x = direction; 1024 threads = 32 waves; wave w<29 owns gate N-tile w
// (Whh^T B tiles register-resident, 5 K-tiles). h kept in LDS bf16 [96][160].
// x_t double-buffered in LDS via async global->LDS copies overlapped with WMMA.
__global__ void __launch_bounds__(1024) rnn_bigru_kernel(
    const float* __restrict__ sent,
    const float* WihF, const float* WhhF, const float* bihF, const float* bhhF,
    const float* WihB, const float* WhhB, const float* bihB, const float* bhhB,
    __bf16* outF, __bf16* outB) {
  extern __shared__ char smem[];
  __bf16* shH  = (__bf16*)(smem);                 // [96][160]      30720 B
  float*  shX0 = (float*)(smem + 30720);          // [96][17] f32    6528 B
  float*  shX1 = (float*)(smem + 37248);          //                 6528 B
  __bf16* shGX = (__bf16*)(smem + 43776);         // [464][96] col-major 89088 B
  __bf16* shGH = (__bf16*)(smem + 132864);        //                 89088 B
  float*  shBI = (float*)(smem + 221952);         // [450]
  float*  shBH = (float*)(smem + 223752);         // [450]

  const int dir = blockIdx.x;
  const float* Wih = dir ? WihB : WihF;
  const float* Whh = dir ? WhhB : WhhF;
  const float* bih = dir ? bihB : bihF;
  const float* bhh = dir ? bhhB : bhhF;
  __bf16* out = dir ? outB : outF;

  const int tid = threadIdx.x;
  const int lane = tid & 31;
  const int w = tid >> 5;

  for (int e = tid; e < NB * 160; e += 1024) shH[e] = (__bf16)0.0f;
  for (int e = tid; e < G1; e += 1024) { shBI[e] = bih[e]; shBH[e] = bhh[e]; }

  v16bf bWih = {};
  v16bf bWhh[5];
  if (w < NT1) {
    bWih = load_b_w(Wih, w * 16, G1, 0, DIN, DIN, lane);
#pragma unroll
    for (int kt = 0; kt < 5; ++kt)
      bWhh[kt] = load_b_w(Whh, w * 16, G1, kt * 32, H1, H1, lane);
  } else {
#pragma unroll
    for (int kt = 0; kt < 5; ++kt) bWhh[kt] = bWih;
  }

  // prologue: async-stage x for the first timestep into buffer 0
  {
    const int t0 = dir ? (T_LEN - 1) : 0;
    unsigned lb = lds_addr32(shX0);
    unsigned long long gb = (unsigned long long)(size_t)(sent + (size_t)t0 * (NB * DIN));
    for (int e = tid; e < NB * DIN; e += 1024) async_g2l_b32(lb + e * 4u, gb, e * 4);
  }

  for (int ts = 0; ts < T_LEN; ++ts) {
    const int tt = dir ? (T_LEN - 1 - ts) : ts;
    const float* xb = (ts & 1) ? shX1 : shX0;
    wait_asynccnt0();          // x_t landed in LDS
    __syncthreads();
    if (ts + 1 < T_LEN) {      // overlap x_{t+1} copy with this step's WMMA
      const int tn = dir ? (T_LEN - 2 - ts) : (ts + 1);
      float* xnb = (ts & 1) ? shX0 : shX1;
      unsigned lb = lds_addr32(xnb);
      unsigned long long gb =
          (unsigned long long)(size_t)(sent + (size_t)tn * (NB * DIN));
      for (int e = tid; e < NB * DIN; e += 1024) async_g2l_b32(lb + e * 4u, gb, e * 4);
    }
    if (w < NT1) {
#pragma unroll
      for (int mt = 0; mt < 6; ++mt) {
        int r0 = mt * 16;
        v8f gx = {};
        gx = wmma_bf(load_a_ldsf32(xb, r0, DIN, DIN, lane), bWih, gx);
        store_c_col_bf(shGX, NB, w * 16, r0, gx, lane);
        v8f gh = {};
#pragma unroll
        for (int kt = 0; kt < 5; ++kt)
          gh = wmma_bf(load_a_lds(shH, r0, 160, kt * 32, lane), bWhh[kt], gh);
        store_c_col_bf(shGH, NB, w * 16, r0, gh, lane);
      }
    }
    __syncthreads();
    for (int e = tid; e < NB * H1; e += 1024) {          // GRU cell update
      int b = e / H1, u = e - b * H1;
      float xr = (float)shGX[u * NB + b]            + shBI[u];
      float xz = (float)shGX[(H1 + u) * NB + b]     + shBI[H1 + u];
      float xn = (float)shGX[(2 * H1 + u) * NB + b] + shBI[2 * H1 + u];
      float hr = (float)shGH[u * NB + b]            + shBH[u];
      float hz = (float)shGH[(H1 + u) * NB + b]     + shBH[H1 + u];
      float hn = (float)shGH[(2 * H1 + u) * NB + b] + shBH[2 * H1 + u];
      float r = sigm(xr + hr);
      float z = sigm(xz + hz);
      float n = tanhf(xn + r * hn);
      float hold = (float)shH[b * 160 + u];
      float hnew = (1.0f - z) * n + z * hold;
      shH[b * 160 + u] = (__bf16)hnew;
      out[(tt * NB + b) * H1 + u] = (__bf16)hnew;
    }
    __syncthreads();
  }
}

// ------------------------------ rnn2 (H2=12) -------------------------------
__global__ void __launch_bounds__(1024) rnn2_bigru_kernel(
    const __bf16* rf, const __bf16* rb,
    const float* WihF, const float* WhhF, const float* bihF, const float* bhhF,
    const float* WihB, const float* WhhB, const float* bihB, const float* bhhB,
    float* outF, float* outB) {
  __shared__ __bf16 shH[NB * 32];
  __shared__ float  shGX[G2P * NB];   // col-major [48][96]
  __shared__ float  shGH[G2P * NB];
  __shared__ float  shBI[G2];
  __shared__ float  shBH[G2];

  const int dir = blockIdx.x;
  const float* Wih = dir ? WihB : WihF;
  const float* Whh = dir ? WhhB : WhhF;
  const float* bih = dir ? bihB : bihF;
  const float* bhh = dir ? bhhB : bhhF;
  float* out = dir ? outB : outF;

  const int tid = threadIdx.x, lane = tid & 31, w = tid >> 5;
  for (int e = tid; e < NB * 32; e += 1024) shH[e] = (__bf16)0.0f;
  if (tid < G2) { shBI[tid] = bih[tid]; shBH[tid] = bhh[tid]; }

  const bool act = (w < 18);
  const int mt = w / 3, nt = w % 3;
  v16bf bX[10];
  v16bf bH = {};
  if (act) {
#pragma unroll
    for (int seg = 0; seg < 2; ++seg)
#pragma unroll
      for (int kt = 0; kt < 5; ++kt)
        bX[seg * 5 + kt] =
            load_b_w(Wih + seg * H1, nt * 16, G2, kt * 32, H1, 2 * H1, lane);
    bH = load_b_w(Whh, nt * 16, G2, 0, H2, H2, lane);
  } else {
#pragma unroll
    for (int i = 0; i < 10; ++i) bX[i] = bH;
  }
  __syncthreads();

  for (int ts = 0; ts < T_LEN; ++ts) {
    const int tt = dir ? (T_LEN - 1 - ts) : ts;
    if (ts + 1 < T_LEN) {   // prefetch next timestep's rnn_out rows into L2/L0
      const int tn = dir ? (T_LEN - 2 - ts) : (ts + 1);
      const __bf16* pf = rf + (size_t)tn * NB * H1;
      const __bf16* pb = rb + (size_t)tn * NB * H1;
      for (int e = tid; e < 450; e += 1024) {   // 450 x 64B lines = 28800 B
        __builtin_prefetch(pf + e * 32, 0, 0);
        __builtin_prefetch(pb + e * 32, 0, 0);
      }
    }
    if (act) {
      int r0 = mt * 16;
      v8f gx = {};
#pragma unroll
      for (int kt = 0; kt < 5; ++kt)
        gx = wmma_bf(load_a_gbf(rf, tt * NB + r0, H1, kt * 32, H1, lane),
                     bX[kt], gx);
#pragma unroll
      for (int kt = 0; kt < 5; ++kt)
        gx = wmma_bf(load_a_gbf(rb, tt * NB + r0, H1, kt * 32, H1, lane),
                     bX[5 + kt], gx);
      store_c_col_f32(shGX, NB, nt * 16, r0, gx, lane);
      v8f gh = {};
      gh = wmma_bf(load_a_lds(shH, r0, 32, 0, lane), bH, gh);
      store_c_col_f32(shGH, NB, nt * 16, r0, gh, lane);
    }
    __syncthreads();
    for (int e = tid; e < NB * H2; e += 1024) {
      int b = e / H2, u = e - b * H2;
      float xr = shGX[u * NB + b]            + shBI[u];
      float xz = shGX[(H2 + u) * NB + b]     + shBI[H2 + u];
      float xn = shGX[(2 * H2 + u) * NB + b] + shBI[2 * H2 + u];
      float hr = shGH[u * NB + b]            + shBH[u];
      float hz = shGH[(H2 + u) * NB + b]     + shBH[H2 + u];
      float hn = shGH[(2 * H2 + u) * NB + b] + shBH[2 * H2 + u];
      float r = sigm(xr + hr);
      float z = sigm(xz + hz);
      float n = tanhf(xn + r * hn);
      float hold = (float)shH[b * 32 + u];
      float hnew = (1.0f - z) * n + z * hold;
      shH[b * 32 + u] = (__bf16)hnew;
      out[(tt * NB + b) * H2 + u] = hnew;
    }
    __syncthreads();
  }
}

// --------------------------- hier1 biGRU (HH=16) ---------------------------
// One wave per 16-row batch tile of the 6144-wide segment batch; the 3 gate
// N-tiles align exactly with r/z/n.  48 blocks x 8 waves = 384 tiles.
__global__ void __launch_bounds__(256) hier1_kernel(
    const float* __restrict__ sent,
    const float* WihF, const float* WhhF, const float* bihF, const float* bhhF,
    const float* WihB, const float* WhhB, const float* bihB, const float* bhhB,
    float* r1) {
  __shared__ __bf16 shH[8 * 16 * 32];
  const int tid = threadIdx.x, lane = tid & 31, w = tid >> 5;
  const int mtile = blockIdx.x * 8 + w;
  const int bn0 = mtile * 16;
  __bf16* hb = shH + w * 512;
  const int nl = lane & 15, hi = (lane >> 4) & 1;

  for (int dir = 0; dir < 2; ++dir) {
    const float* Wih = dir ? WihB : WihF;
    const float* Whh = dir ? WhhB : WhhF;
    const float* bih = dir ? bihB : bihF;
    const float* bhh = dir ? bhhB : bhhF;
    for (int e = lane; e < 512; e += 32) hb[e] = (__bf16)0.0f;
    v16bf bI[3], bHt[3];
    float bi[3], bh[3];
#pragma unroll
    for (int g = 0; g < 3; ++g) {
      bI[g]  = load_b_w(Wih, g * 16, GH_, 0, DIN, DIN, lane);
      bHt[g] = load_b_w(Whh, g * 16, GH_, 0, HH, HH, lane);
      bi[g] = bih[g * 16 + nl];
      bh[g] = bhh[g * 16 + nl];
    }
    for (int sj = 0; sj < SEG; ++sj) {
      const int step = dir ? (SEG - 1 - sj) : sj;
      v16bf ax;
      {
        int bn = bn0 + (lane & 15);
        int b = bn >> 6, s = bn & 63;
        int t = s * 16 + step;
        const float* xp = sent + (t * NB + b) * DIN;
#pragma unroll
        for (int i = 0; i < 16; ++i) {
          int k = a_kidx(i, hi);
          ax[i] = (k < DIN) ? (__bf16)xp[k] : (__bf16)0.0f;
        }
      }
      v16bf ah = load_a_lds(hb, 0, 32, 0, lane);
      v8f gx0 = {}, gx1 = {}, gx2 = {}, gh0 = {}, gh1 = {}, gh2 = {};
      gx0 = wmma_bf(ax, bI[0], gx0);
      gx1 = wmma_bf(ax, bI[1], gx1);
      gx2 = wmma_bf(ax, bI[2], gx2);
      gh0 = wmma_bf(ah, bHt[0], gh0);
      gh1 = wmma_bf(ah, bHt[1], gh1);
      gh2 = wmma_bf(ah, bHt[2], gh2);
#pragma unroll
      for (int v = 0; v < 8; ++v) {
        int m = hi * 8 + v;
        float r = sigm(gx0[v] + bi[0] + gh0[v] + bh[0]);
        float z = sigm(gx1[v] + bi[1] + gh1[v] + bh[1]);
        float n = tanhf(gx2[v] + bi[2] + r * (gh2[v] + bh[2]));
        float hold = (float)hb[m * 32 + nl];
        float hnew = (1.0f - z) * n + z * hold;
        hb[m * 32 + nl] = (__bf16)hnew;
        r1[(step * BN + bn0 + m) * 32 + dir * 16 + nl] = hnew;
      }
    }
  }
}

// --------------------- hier2: 1-step biGRU over ctx1 -----------------------
__global__ void __launch_bounds__(256) hier2_kernel(
    const float* __restrict__ ctx1,
    const float* WihF, const float* bihF, const float* bhhF,
    const float* WihB, const float* bihB, const float* bhhB, float* r2) {
  const int tid = threadIdx.x, lane = tid & 31, w = tid >> 5;
  const int mtile = blockIdx.x * 8 + w;   // 0..95 over 1536 rows
  const int r0 = mtile * 16;
  const int nl = lane & 15, hi = (lane >> 4) & 1;

  v16bf ax;   // K = 32 exactly
  {
    int m = r0 + (lane & 15);
#pragma unroll
    for (int i = 0; i < 16; ++i) ax[i] = (__bf16)ctx1[m * 32 + a_kidx(i, hi)];
  }
  for (int dir = 0; dir < 2; ++dir) {
    const float* Wih = dir ? WihB : WihF;
    const float* bih = dir ? bihB : bihF;
    const float* bhh = dir ? bhhB : bhhF;
    v16bf bW[3];
    float bi[3], bh[3];
#pragma unroll
    for (int g = 0; g < 3; ++g) {
      bW[g] = load_b_w(Wih, g * 16, GH_, 0, 32, 32, lane);
      bi[g] = bih[g * 16 + nl];
      bh[g] = bhh[g * 16 + nl];
    }
    v8f g0 = {}, g1 = {}, g2 = {};
    g0 = wmma_bf(ax, bW[0], g0);
    g1 = wmma_bf(ax, bW[1], g1);
    g2 = wmma_bf(ax, bW[2], g2);
#pragma unroll
    for (int v = 0; v < 8; ++v) {
      int m = hi * 8 + v;
      float r = sigm(g0[v] + bi[0] + bh[0]);   // h0 = 0 -> gh = bhh
      float z = sigm(g1[v] + bi[1] + bh[1]);
      float n = tanhf(g2[v] + bi[2] + r * bh[2]);
      r2[(r0 + m) * 32 + dir * 16 + nl] = (1.0f - z) * n;
    }
  }
}

// ------------------------- dot-product attention ---------------------------
// mode 0: att1 (block q=b*16+j, rows r1[j, b*64 + s, :], L=64)
// mode 1: att2 (block b, rows contiguous, L=16)
__global__ void __launch_bounds__(256) attn_kernel(
    const float* __restrict__ X, const float* Wq, const float* bq,
    const float* Wv, const float* bv, float* out, int L, int mode) {
  __shared__ float sx[64 * 32];
  __shared__ float sq[64 * 32];
  __shared__ float sv[64 * 32];
  __shared__ float ss[64 * 64];
  __shared__ float sw[64];
  const int q = blockIdx.x, tid = threadIdx.x;
  const float* base;
  if (mode == 0) {
    int b = q >> 4, j = q & 15;
    base = X + (j * BN + b * NSEG) * 32;
  } else {
    base = X + q * (L * 32);
  }
  for (int e = tid; e < L * 32; e += 256) sx[e] = base[e];
  __syncthreads();
  for (int e = tid; e < L * 32; e += 256) {
    int l = e >> 5, h = e & 31;
    float aq = bq[h], av = bv[h];
    for (int k = 0; k < 32; ++k) {
      float x = sx[l * 32 + k];
      aq += x * Wq[h * 32 + k];
      av += x * Wv[h * 32 + k];
    }
    sq[e] = aq;
    sv[e] = av;
  }
  __syncthreads();
  for (int e = tid; e < L * L; e += 256) {
    int l = e / L, m = e - l * L;
    float a = 0.0f;
    for (int h = 0; h < 32; ++h) a += sq[l * 32 + h] * sv[m * 32 + h];
    ss[l * 64 + m] = a;
  }
  __syncthreads();
  if (tid < L) {
    int l = tid;
    float mx = -1e30f;
    for (int m = 0; m < L; ++m) mx = fmaxf(mx, ss[l * 64 + m]);
    float s = 0.0f;
    for (int m = 0; m < L; ++m) {
      float e2 = __expf(ss[l * 64 + m] - mx);
      ss[l * 64 + m] = e2;
      s += e2;
    }
    float inv = 1.0f / s;
    for (int m = 0; m < L; ++m) ss[l * 64 + m] *= inv;
  }
  __syncthreads();
  if (tid < L) {
    int m = tid;
    float s = 0.0f;
    for (int l = 0; l < L; ++l) s += ss[l * 64 + m];
    sw[m] = s;
  }
  __syncthreads();
  if (tid < 32) {
    int h = tid;
    float a = 0.0f;
    for (int m = 0; m < L; ++m) a += sw[m] * sv[m * 32 + h];
    out[q * 32 + h] = a;
  }
}

// ------------------------------ CE losses ----------------------------------
__global__ void __launch_bounds__(64) loss_pitch_kernel(
    const __bf16* rf, const __bf16* rb, const float* hier, const float* Wp,
    const float* bp, const int* tgt, float* acc) {
  __shared__ float sx[DPC];
  __shared__ float slg[NP];
  const int i = blockIdx.x;
  const int t = i / NB, b = i - t * NB;
  const int tid = threadIdx.x;
  for (int k = tid; k < DPC; k += 64) {
    float v;
    if (k < H1) v = (float)rf[(t * NB + b) * H1 + k];
    else if (k < 2 * H1) v = (float)rb[(t * NB + b) * H1 + (k - H1)];
    else v = hier[b * 32 + (k - 2 * H1)];
    sx[k] = v;
  }
  __syncthreads();
  if (tid < NP) {
    float a = bp[tid];
    const float* wr = Wp + tid * DPC;
    for (int k = 0; k < DPC; ++k) a += sx[k] * wr[k];
    slg[tid] = a;
  }
  __syncthreads();
  if (tid == 0) {
    int tg = tgt[i];
    if (tg != NP - 1) {
      float mx = -1e30f;
      for (int c = 0; c < NP; ++c) mx = fmaxf(mx, slg[c]);
      float s = 0.0f;
      for (int c = 0; c < NP; ++c) s += __expf(slg[c] - mx);
      float nll = (mx + __logf(s)) - slg[tg];
      atomicAdd(&acc[0], nll);
      atomicAdd(&acc[1], 1.0f);
    }
  }
}

__global__ void __launch_bounds__(64) loss_ks_kernel(
    const float* r2f, const float* r2b, const float* hier, const float* Wk,
    const float* bk, const int* tgt, float* acc) {
  __shared__ float sx[DKC];
  __shared__ float slg[NK];
  const int i = blockIdx.x;
  const int t = i / NB, b = i - t * NB;
  const int tid = threadIdx.x;
  for (int k = tid; k < DKC; k += 64) {
    float v;
    if (k < H2) v = r2f[(t * NB + b) * H2 + k];
    else if (k < 2 * H2) v = r2b[(t * NB + b) * H2 + (k - H2)];
    else v = hier[b * 32 + (k - 2 * H2)];
    sx[k] = v;
  }
  __syncthreads();
  if (tid < NK) {
    float a = bk[tid];
    const float* wr = Wk + tid * DKC;
    for (int k = 0; k < DKC; ++k) a += sx[k] * wr[k];
    slg[tid] = a;
  }
  __syncthreads();
  if (tid == 0) {
    int tg = tgt[i];
    if (tg != NK - 1) {
      float mx = -1e30f;
      for (int c = 0; c < NK; ++c) mx = fmaxf(mx, slg[c]);
      float s = 0.0f;
      for (int c = 0; c < NK; ++c) s += __expf(slg[c] - mx);
      float nll = (mx + __logf(s)) - slg[tg];
      atomicAdd(&acc[2], nll);
      atomicAdd(&acc[3], 1.0f);
    }
  }
}

__global__ void init_kernel(float* acc) {
  if (threadIdx.x < 4) acc[threadIdx.x] = 0.0f;
}

__global__ void finalize_kernel(const float* acc, float* out) {
  if (threadIdx.x == 0) {
    out[0] = acc[0] / fmaxf(acc[1], 1.0f) + acc[2] / fmaxf(acc[3], 1.0f);
  }
}

// ------------------------------- launcher ----------------------------------
extern "C" void kernel_launch(void* const* d_in, const int* in_sizes, int n_in,
                              void* d_out, int out_size, void* d_ws,
                              size_t ws_size, hipStream_t stream) {
  (void)in_sizes; (void)n_in; (void)out_size; (void)ws_size;
  const float* sent = (const float*)d_in[0];
  // rnn
  const float* rWihF = (const float*)d_in[1];
  const float* rWhhF = (const float*)d_in[2];
  const float* rbihF = (const float*)d_in[3];
  const float* rbhhF = (const float*)d_in[4];
  const float* rWihB = (const float*)d_in[5];
  const float* rWhhB = (const float*)d_in[6];
  const float* rbihB = (const float*)d_in[7];
  const float* rbhhB = (const float*)d_in[8];
  // rnn2
  const float* r2WihF = (const float*)d_in[9];
  const float* r2WhhF = (const float*)d_in[10];
  const float* r2bihF = (const float*)d_in[11];
  const float* r2bhhF = (const float*)d_in[12];
  const float* r2WihB = (const float*)d_in[13];
  const float* r2WhhB = (const float*)d_in[14];
  const float* r2bihB = (const float*)d_in[15];
  const float* r2bhhB = (const float*)d_in[16];
  // hier1
  const float* h1WihF = (const float*)d_in[17];
  const float* h1WhhF = (const float*)d_in[18];
  const float* h1bihF = (const float*)d_in[19];
  const float* h1bhhF = (const float*)d_in[20];
  const float* h1WihB = (const float*)d_in[21];
  const float* h1WhhB = (const float*)d_in[22];
  const float* h1bihB = (const float*)d_in[23];
  const float* h1bhhB = (const float*)d_in[24];
  // hier2 (Whh unused: h0 == 0)
  const float* h2WihF = (const float*)d_in[25];
  const float* h2bihF = (const float*)d_in[27];
  const float* h2bhhF = (const float*)d_in[28];
  const float* h2WihB = (const float*)d_in[29];
  const float* h2bihB = (const float*)d_in[31];
  const float* h2bhhB = (const float*)d_in[32];
  // attention
  const float* a1Wq = (const float*)d_in[33];
  const float* a1bq = (const float*)d_in[34];
  const float* a1Wv = (const float*)d_in[35];
  const float* a1bv = (const float*)d_in[36];
  const float* a2Wq = (const float*)d_in[37];
  const float* a2bq = (const float*)d_in[38];
  const float* a2Wv = (const float*)d_in[39];
  const float* a2bv = (const float*)d_in[40];
  const float* Wp = (const float*)d_in[41];
  const float* bp = (const float*)d_in[42];
  const float* Wk = (const float*)d_in[43];
  const float* bk = (const float*)d_in[44];
  const int* pitches = (const int*)d_in[45];
  const int* keysigs = (const int*)d_in[46];

  char* ws = (char*)d_ws;
  __bf16* rnnF  = (__bf16*)(ws + 0);           // 1024*96*150 bf16
  __bf16* rnnB  = (__bf16*)(ws + 29491200);
  float*  r2F   = (float*)(ws + 58982400);     // 1024*96*12 f32
  float*  r2B   = (float*)(ws + 63700992);
  float*  r1    = (float*)(ws + 68419584);     // 16*6144*32 f32
  float*  ctx1  = (float*)(ws + 81002496);     // 1536*32
  float*  r2m   = (float*)(ws + 81199104);     // 1536*32
  float*  hier  = (float*)(ws + 81395712);     // 96*32
  float*  lacc  = (float*)(ws + 81408000);     // 4 floats

  init_kernel<<<1, 32, 0, stream>>>(lacc);

  // hierarchical context path
  hier1_kernel<<<48, 256, 0, stream>>>(sent, h1WihF, h1WhhF, h1bihF, h1bhhF,
                                       h1WihB, h1WhhB, h1bihB, h1bhhB, r1);
  attn_kernel<<<1536, 256, 0, stream>>>(r1, a1Wq, a1bq, a1Wv, a1bv, ctx1, 64, 0);
  hier2_kernel<<<12, 256, 0, stream>>>(ctx1, h2WihF, h2bihF, h2bhhF, h2WihB,
                                       h2bihB, h2bhhB, r2m);
  attn_kernel<<<96, 256, 0, stream>>>(r2m, a2Wq, a2bq, a2Wv, a2bv, hier, 16, 1);

  // main recurrent path (both directions concurrent as blocks 0/1)
  rnn_bigru_kernel<<<2, 1024, 225552, stream>>>(sent, rWihF, rWhhF, rbihF,
                                                rbhhF, rWihB, rWhhB, rbihB,
                                                rbhhB, rnnF, rnnB);
  rnn2_bigru_kernel<<<2, 1024, 0, stream>>>(rnnF, rnnB, r2WihF, r2WhhF, r2bihF,
                                            r2bhhF, r2WihB, r2WhhB, r2bihB,
                                            r2bhhB, r2F, r2B);

  // classifier heads + masked-mean CE
  loss_pitch_kernel<<<T_LEN * NB, 64, 0, stream>>>(rnnF, rnnB, hier, Wp, bp,
                                                   pitches, lacc);
  loss_ks_kernel<<<T_LEN * NB, 64, 0, stream>>>(r2F, r2B, hier, Wk, bk,
                                                keysigs, lacc);
  finalize_kernel<<<1, 32, 0, stream>>>(lacc, (float*)d_out);
}
